// MPC_11269994184825
// MI455X (gfx1250) — matrix operand order repaired and checked by reading, not compiled, not run on hardware
//
#include <hip/hip_runtime.h>

// ---------------------------------------------------------------------------
// CMT block forward for MI455X (gfx1250, wave32, WMMA f32_16x16x32_f16,
// async-to-LDS staging for GEMM A tiles)
// ---------------------------------------------------------------------------

typedef __attribute__((ext_vector_type(16))) _Float16 v16h;
typedef __attribute__((ext_vector_type(8)))  float    v8f;

union AFrag { v16h v; _Float16 e[16]; uint4 q[2]; };
union CFrag { v8f  v; float    e[8];  };

__device__ inline v8f wmma32f16(v16h a, v16h b, v8f c) {
  // D = A(16x32 f16) * B(32x16 f16) + C(16x16 f32)
  return __builtin_amdgcn_wmma_f32_16x16x32_f16(false, a, false, b,
                                                (short)0, c, false, false);
}

__device__ inline float gelu_exact(float x) {
  return 0.5f * x * (1.0f + erff(x * 0.7071067811865475f));
}

#define BQ    8
#define CCH   256
#define HH    56
#define WWW   56
#define NTOK  3136
#define NKVT  784
#define HDIM  64
#define HIDC  1024

// ---------------------------------------------------------------------------
// Kernel 1: local perception unit: x1 = dwconv3x3(x) + x     (fp32 -> fp32)
// ---------------------------------------------------------------------------
__global__ void __launch_bounds__(256) lpu_kernel(
    const float* __restrict__ x, const float* __restrict__ dww,
    const float* __restrict__ dwb, float* __restrict__ x1) {
  int b = blockIdx.x / HH;
  int h = blockIdx.x % HH;
  int c = threadIdx.x;
  float w[9];
#pragma unroll
  for (int i = 0; i < 9; ++i) w[i] = dww[c * 9 + i];
  float bias = dwb[c];
  const float* xb = x  + ((size_t)b * NTOK) * CCH + c;
  float*       ob = x1 + ((size_t)b * NTOK) * CCH + c;
  for (int wc = 0; wc < WWW; ++wc) {
    float acc = bias;
#pragma unroll
    for (int dh = -1; dh <= 1; ++dh) {
      int hh = h + dh;
      if (hh < 0 || hh >= HH) continue;
#pragma unroll
      for (int dw = -1; dw <= 1; ++dw) {
        int wcc = wc + dw;
        if (wcc < 0 || wcc >= WWW) continue;
        acc += xb[(size_t)(hh * WWW + wcc) * CCH] * w[(dh + 1) * 3 + (dw + 1)];
      }
    }
    int n = h * WWW + wc;
    ob[(size_t)n * CCH] = acc + xb[(size_t)n * CCH];
  }
}

// ---------------------------------------------------------------------------
// Kernel 2: row LayerNorm over C=256, one wave per token, eps=1e-6  -> f16
// ---------------------------------------------------------------------------
__global__ void __launch_bounds__(256) ln_kernel(
    const float* __restrict__ xin, const float* __restrict__ g,
    const float* __restrict__ be, _Float16* __restrict__ out) {
  int wave = threadIdx.x >> 5;
  int lane = threadIdx.x & 31;
  size_t tok = (size_t)blockIdx.x * 8 + wave;
  const float* row = xin + tok * CCH + lane * 8;
  const float4* r4 = reinterpret_cast<const float4*>(row);
  float4 p0 = r4[0], p1 = r4[1];
  float v[8] = {p0.x, p0.y, p0.z, p0.w, p1.x, p1.y, p1.z, p1.w};
  float s = 0.f, sq = 0.f;
#pragma unroll
  for (int i = 0; i < 8; ++i) { s += v[i]; sq += v[i] * v[i]; }
#pragma unroll
  for (int off = 16; off > 0; off >>= 1) {
    s  += __shfl_xor(s,  off, 32);
    sq += __shfl_xor(sq, off, 32);
  }
  float mean = s * (1.f / 256.f);
  float var  = sq * (1.f / 256.f) - mean * mean;
  float rstd = rsqrtf(var + 1e-6f);
  _Float16* orow = out + tok * CCH + lane * 8;
#pragma unroll
  for (int i = 0; i < 8; ++i)
    orow[i] = (_Float16)((v[i] - mean) * rstd * g[lane * 8 + i] + be[lane * 8 + i]);
}

// ---------------------------------------------------------------------------
// Kernel 3: SR path: y = BN(dwconv2x2 stride2 (xn))   f16 in -> f16 out
// ---------------------------------------------------------------------------
__global__ void __launch_bounds__(256) srbn_kernel(
    const _Float16* __restrict__ xn, const float* __restrict__ srw,
    const float* __restrict__ srb, const float* __restrict__ g,
    const float* __restrict__ bb, const float* __restrict__ mm,
    const float* __restrict__ vv, _Float16* __restrict__ y) {
  int b  = blockIdx.x / 28;
  int h2 = blockIdx.x % 28;
  int c  = threadIdx.x;
  float w00 = srw[c * 4 + 0], w01 = srw[c * 4 + 1];
  float w10 = srw[c * 4 + 2], w11 = srw[c * 4 + 3];
  float sc = g[c] * rsqrtf(vv[c] + 1e-5f);
  float bias = srb[c], bmean = mm[c], bbeta = bb[c];
  const _Float16* xb = xn + ((size_t)b * NTOK) * CCH + c;
  for (int w2 = 0; w2 < 28; ++w2) {
    int h = 2 * h2, w = 2 * w2;
    float a = (float)xb[(size_t)(h * WWW + w) * CCH] * w00
            + (float)xb[(size_t)(h * WWW + w + 1) * CCH] * w01
            + (float)xb[(size_t)((h + 1) * WWW + w) * CCH] * w10
            + (float)xb[(size_t)((h + 1) * WWW + w + 1) * CCH] * w11 + bias;
    y[((size_t)(b * NKVT + h2 * 28 + w2)) * CCH + c] =
        (_Float16)((a - bmean) * sc + bbeta);
  }
}

// ---------------------------------------------------------------------------
// Kernel 4: tiled WMMA GEMM, 128x64 tile per 256-thread block (8 waves).
//  * A tile staged to LDS via GLOBAL_LOAD_ASYNC_TO_LDS_B128 (ASYNCcnt path)
//  * W tile staged fp32->f16 in fragment-contiguous layout so every WMMA
//    B-fragment is two ds_load_b128 (no strided 16-bit LDS reads)
//   WTRANS=false: W[k][n] = Wm[k*N + n]     (q_w/k_w/v_w/ao_w layout)
//   WTRANS=true : W[k][n] = Wm[n*K + k]     (1x1 conv weight [N,K] layout)
// ---------------------------------------------------------------------------
constexpr int EP_F16          = 0;  // f16 out
constexpr int EP_RESID_F32    = 1;  // fp32 out = resid + gemm
constexpr int EP_GELU_BN_F16  = 2;  // f16 out = BN(gelu(gemm))
constexpr int EP_BN_RESID_F32 = 3;  // fp32 out = resid + BN(gemm)

template <int EP, bool WTRANS>
__global__ void __launch_bounds__(256) gemm_kernel(
    const _Float16* __restrict__ A, const float* __restrict__ Wm,
    const float* __restrict__ bias, const float* __restrict__ resid,
    const float* __restrict__ bng, const float* __restrict__ bnb,
    const float* __restrict__ bnm, const float* __restrict__ bnv,
    void* __restrict__ outp, int K, int Nn) {
  __shared__ __align__(16) _Float16 As[128 * 32];
  __shared__ __align__(16) _Float16 Ws[2048];  // fragment-contiguous W tile
  int t    = threadIdx.x;
  int wv   = t >> 5;
  int lane = t & 31;
  int m0 = blockIdx.x * 128, n0 = blockIdx.y * 64;

  CFrag acc[4];
#pragma unroll
  for (int i = 0; i < 4; ++i)
#pragma unroll
    for (int r = 0; r < 8; ++r) acc[i].e[r] = 0.f;

  int ar = t >> 1, ap = t & 1;                 // 2 threads per A row
  const _Float16* asrc = A + (size_t)(m0 + ar) * K + ap * 16;
  // generic pointers to __shared__ carry the LDS byte offset in bits [31:0]
  unsigned lds0 = (unsigned)(size_t)(void*)&As[ar * 32 + ap * 16];

  for (int k0 = 0; k0 < K; k0 += 32) {
    // --- async stage A tile [128 x 32] f16: 2 x b128 per thread -----------
    const _Float16* g0 = asrc + k0;
    asm volatile(
        "global_load_async_to_lds_b128 %0, %2, off\n\t"
        "global_load_async_to_lds_b128 %1, %3, off"
        :
        : "v"(lds0), "v"(lds0 + 16u),
          "v"((unsigned long long)(size_t)g0),
          "v"((unsigned long long)(size_t)(g0 + 8))
        : "memory");
    if (k0 + 32 < K) __builtin_prefetch(asrc + k0 + 32, 0, 1);

    // --- stage W tile [32 x 64] fp32 -> f16, fragment-contiguous ----------
    // Ws[(((sub*2 + kg)*16 + n)*16) + j] = W[k0 + kg*16 + j][n0 + sub*16 + n]
#pragma unroll
    for (int i = 0; i < 8; ++i) {
      int idx = t * 8 + i;
      int j   = idx & 15;
      int n   = (idx >> 4) & 15;
      int kg  = (idx >> 8) & 1;
      int sub = idx >> 9;
      float wval;
      if (!WTRANS)
        wval = Wm[(size_t)(k0 + kg * 16 + j) * Nn + n0 + sub * 16 + n];
      else
        wval = Wm[(size_t)(n0 + sub * 16 + n) * K + k0 + kg * 16 + j];
      Ws[idx] = (_Float16)wval;
    }
    asm volatile("s_wait_asynccnt 0x0" ::: "memory");
    __syncthreads();

    // --- A fragment (16x32): two contiguous 16B runs per lane -------------
    int am = wv * 16 + (lane & 15);
    int kb = (lane >> 4) * 16;
    AFrag au;
    const uint4* a16 = reinterpret_cast<const uint4*>(&As[am * 32 + kb]);
    au.q[0] = a16[0];   // halves kb .. kb+7
    au.q[1] = a16[1];   // halves kb+8 .. kb+15
#pragma unroll
    for (int sub = 0; sub < 4; ++sub) {
      AFrag bu;
      int fbase = ((sub * 2 + (lane >> 4)) * 16 + (lane & 15)) * 16;
      const uint4* b16 = reinterpret_cast<const uint4*>(&Ws[fbase]);
      bu.q[0] = b16[0];
      bu.q[1] = b16[1];
      acc[sub].v = wmma32f16(au.v, bu.v, acc[sub].v);
    }
    __syncthreads();
  }

  // epilogue: C/D layout: row = r + 8*(lane>>4), col = lane&15
#pragma unroll
  for (int sub = 0; sub < 4; ++sub) {
#pragma unroll
    for (int r = 0; r < 8; ++r) {
      int row = m0 + wv * 16 + r + 8 * (lane >> 4);
      int col = n0 + sub * 16 + (lane & 15);
      size_t idx = (size_t)row * Nn + col;
      float tv = acc[sub].e[r] + bias[col];
      if (EP == EP_F16) {
        ((_Float16*)outp)[idx] = (_Float16)tv;
      } else if (EP == EP_RESID_F32) {
        ((float*)outp)[idx] = resid[idx] + tv;
      } else if (EP == EP_GELU_BN_F16) {
        float gl = gelu_exact(tv);
        float sc = bng[col] * rsqrtf(bnv[col] + 1e-5f);
        ((_Float16*)outp)[idx] = (_Float16)((gl - bnm[col]) * sc + bnb[col]);
      } else {
        float sc = bng[col] * rsqrtf(bnv[col] + 1e-5f);
        ((float*)outp)[idx] = resid[idx] + (tv - bnm[col]) * sc + bnb[col];
      }
    }
  }
}

// ---------------------------------------------------------------------------
// Kernel 5: attention, one wave per (b, head, 16-row q tile).
//   S = Q Kt * scale + relpos  (WMMA, scores in LDS fp32 16x800, cols padded)
//   softmax rows in LDS (2 lanes/row), P kept fp32 in-place
//   O = P V  (WMMA, P converted to f16 at fragment load)
// o may alias q: each workgroup reads exactly the q region it later writes.
// ---------------------------------------------------------------------------
__global__ void __launch_bounds__(32) attn_kernel(
    const _Float16* __restrict__ q, const _Float16* __restrict__ k,
    const _Float16* __restrict__ v, const float* __restrict__ rp,
    _Float16* __restrict__ o) {
  __shared__ __align__(16) float S[16][800];
  int qt = blockIdx.x, hd = blockIdx.y, b = blockIdx.z;
  int lane = threadIdx.x;
  const float scale = 0.125f;  // 64^-0.5
  int m   = lane & 15;
  int kb  = (lane >> 4) * 8;
  int nn  = lane & 15;
  int kb2 = (lane >> 4) * 16;

  const _Float16* qrow = q + ((size_t)(b * NTOK + qt * 16 + m)) * CCH + hd * HDIM;
  AFrag a0, a1;
#pragma unroll
  for (int j = 0; j < 8; ++j) {
    a0.e[j]     = qrow[kb2 + j];
    a0.e[8 + j] = qrow[kb2 + 8 + j];
    a1.e[j]     = qrow[32 + kb2 + j];
    a1.e[8 + j] = qrow[32 + kb2 + 8 + j];
  }

  const float* rpb = rp + ((size_t)hd * NTOK + (size_t)qt * 16) * NKVT;
  for (int kt = 0; kt < 49; ++kt) {
    const _Float16* krow =
        k + ((size_t)(b * NKVT + kt * 16 + nn)) * CCH + hd * HDIM;
    AFrag b0, b1;
#pragma unroll
    for (int j = 0; j < 16; ++j) {
      b0.e[j] = krow[kb2 + j];        // Kt[kdim][n] = K[token n][kdim]
      b1.e[j] = krow[32 + kb2 + j];
    }
    CFrag c;
#pragma unroll
    for (int r = 0; r < 8; ++r) c.e[r] = 0.f;
    c.v = wmma32f16(a0.v, b0.v, c.v);
    c.v = wmma32f16(a1.v, b1.v, c.v);
#pragma unroll
    for (int r = 0; r < 8; ++r) {
      int mm = r + 8 * (lane >> 4);
      S[mm][kt * 16 + nn] = c.e[r] * scale + rpb[(size_t)mm * NKVT + kt * 16 + nn];
    }
  }
  __syncthreads();

  // softmax: 2 lanes per row, 392 cols each
  int row = lane >> 1, half = lane & 1;
  int c0 = half * 392, c1e = c0 + 392;
  float mx = -3.0e38f;
  for (int cc = c0; cc < c1e; ++cc) mx = fmaxf(mx, S[row][cc]);
  mx = fmaxf(mx, __shfl_xor(mx, 1, 32));
  float sm = 0.f;
  for (int cc = c0; cc < c1e; ++cc) {
    float e = expf(S[row][cc] - mx);
    S[row][cc] = e;
    sm += e;
  }
  sm += __shfl_xor(sm, 1, 32);
  float inv = 1.f / sm;
  for (int cc = c0; cc < c1e; ++cc) S[row][cc] *= inv;
#pragma unroll
  for (int i = 0; i < 8; ++i) S[row][784 + half * 8 + i] = 0.f;  // zero pad
  __syncthreads();

  // O = P @ V : 25 K-chunks of 32 (784 padded to 800), 4 head-dim subtiles
  CFrag oc[4];
#pragma unroll
  for (int i = 0; i < 4; ++i)
#pragma unroll
    for (int r = 0; r < 8; ++r) oc[i].e[r] = 0.f;

  for (int ck = 0; ck < 25; ++ck) {
    AFrag pa;
#pragma unroll
    for (int j = 0; j < 8; ++j) {
      pa.e[j]     = (_Float16)S[m][ck * 32 + kb2 + j];
      pa.e[8 + j] = (_Float16)S[m][ck * 32 + kb2 + 8 + j];
    }
#pragma unroll
    for (int sub = 0; sub < 4; ++sub) {
      AFrag vb;
#pragma unroll
      for (int j = 0; j < 16; ++j) {
        int kv = ck * 32 + kb2 + j;
        vb.e[j] = (kv < NKVT)
            ? v[((size_t)(b * NKVT + kv)) * CCH + hd * HDIM + sub * 16 + nn]
            : (_Float16)0.f;
      }
      oc[sub].v = wmma32f16(pa.v, vb.v, oc[sub].v);
    }
  }
#pragma unroll
  for (int sub = 0; sub < 4; ++sub)
#pragma unroll
    for (int r = 0; r < 8; ++r) {
      int mm = r + 8 * (lane >> 4);
      o[((size_t)(b * NTOK + qt * 16 + mm)) * CCH + hd * HDIM + sub * 16 + nn] =
          (_Float16)oc[sub].e[r];
    }
}

// ---------------------------------------------------------------------------
// Kernel 6: MLP depthwise: h2 = BN2(gelu(dwconv3x3(h) + dw_b + h))  f16->f16
// ---------------------------------------------------------------------------
__global__ void __launch_bounds__(256) dwmlp_kernel(
    const _Float16* __restrict__ hbuf, const float* __restrict__ dww,
    const float* __restrict__ dwb, const float* __restrict__ g2,
    const float* __restrict__ b2, const float* __restrict__ m2,
    const float* __restrict__ v2, _Float16* __restrict__ h2) {
  int b   = blockIdx.x / (HH * 4);
  int rem = blockIdx.x % (HH * 4);
  int h   = rem >> 2;
  int cg  = rem & 3;
  int c   = cg * 256 + threadIdx.x;
  float w[9];
#pragma unroll
  for (int i = 0; i < 9; ++i) w[i] = dww[c * 9 + i];
  float bias = dwb[c];
  float sc = g2[c] * rsqrtf(v2[c] + 1e-5f);
  float bmean = m2[c], bbeta = b2[c];
  const _Float16* hb = hbuf + ((size_t)b * NTOK) * HIDC + c;
  _Float16*       ob = h2   + ((size_t)b * NTOK) * HIDC + c;
  for (int wc = 0; wc < WWW; ++wc) {
    float acc = bias;
#pragma unroll
    for (int dh = -1; dh <= 1; ++dh) {
      int hh = h + dh;
      if (hh < 0 || hh >= HH) continue;
#pragma unroll
      for (int dw = -1; dw <= 1; ++dw) {
        int wcc = wc + dw;
        if (wcc < 0 || wcc >= WWW) continue;
        acc += (float)hb[(size_t)(hh * WWW + wcc) * HIDC] * w[(dh + 1) * 3 + (dw + 1)];
      }
    }
    int n = h * WWW + wc;
    float tv = acc + (float)hb[(size_t)n * HIDC];   // residual before gelu
    tv = gelu_exact(tv);
    ob[(size_t)n * HIDC] = (_Float16)((tv - bmean) * sc + bbeta);
  }
}

// ---------------------------------------------------------------------------
// Host-side launch
// ---------------------------------------------------------------------------
extern "C" void kernel_launch(void* const* d_in, const int* in_sizes, int n_in,
                              void* d_out, int out_size, void* d_ws,
                              size_t ws_size, hipStream_t stream) {
  (void)in_sizes; (void)n_in; (void)out_size; (void)ws_size;
  const float* x    = (const float*)d_in[0];
  const float* rp   = (const float*)d_in[1];
  const float* pdw  = (const float*)d_in[2];
  const float* pdb  = (const float*)d_in[3];
  const float* ln1g = (const float*)d_in[4];
  const float* ln1b = (const float*)d_in[5];
  const float* qw   = (const float*)d_in[6];
  const float* qbi  = (const float*)d_in[7];
  const float* kw   = (const float*)d_in[8];
  const float* kbi  = (const float*)d_in[9];
  const float* vw   = (const float*)d_in[10];
  const float* vbi  = (const float*)d_in[11];
  const float* aow  = (const float*)d_in[12];
  const float* aob  = (const float*)d_in[13];
  const float* srw  = (const float*)d_in[14];
  const float* srb  = (const float*)d_in[15];
  const float* sbg  = (const float*)d_in[16];
  const float* sbb  = (const float*)d_in[17];
  const float* sbm  = (const float*)d_in[18];
  const float* sbv  = (const float*)d_in[19];
  const float* ln2g = (const float*)d_in[20];
  const float* ln2b = (const float*)d_in[21];
  const float* c1w  = (const float*)d_in[22];
  const float* c1b  = (const float*)d_in[23];
  const float* bn1g = (const float*)d_in[24];
  const float* bn1b = (const float*)d_in[25];
  const float* bn1m = (const float*)d_in[26];
  const float* bn1v = (const float*)d_in[27];
  const float* dww  = (const float*)d_in[28];
  const float* dwb  = (const float*)d_in[29];
  const float* bn2g = (const float*)d_in[30];
  const float* bn2b = (const float*)d_in[31];
  const float* bn2m = (const float*)d_in[32];
  const float* bn2v = (const float*)d_in[33];
  const float* c2w  = (const float*)d_in[34];
  const float* c2b  = (const float*)d_in[35];
  const float* bn3g = (const float*)d_in[36];
  const float* bn3b = (const float*)d_in[37];
  const float* bn3m = (const float*)d_in[38];
  const float* bn3v = (const float*)d_in[39];

  // workspace layout (aliasing: x2->x1, xn2->xn, o->q)
  constexpr size_t SZ_X1 = (size_t)BQ * NTOK * CCH * 4;   // 25,690,112
  constexpr size_t SZ_XN = (size_t)BQ * NTOK * CCH * 2;   // 12,845,056
  constexpr size_t SZ_Y  = (size_t)BQ * NKVT * CCH * 2;   //  3,211,264
  constexpr size_t SZ_H  = (size_t)BQ * NTOK * HIDC * 2;  // 51,380,224
  char* ws = (char*)d_ws;
  float*    x1 = (float*)(ws);                                    // + x2
  _Float16* xn = (_Float16*)(ws + SZ_X1);                         // + xn2
  _Float16* qb = (_Float16*)(ws + SZ_X1 + SZ_XN);                 // + o
  _Float16* yb = (_Float16*)(ws + SZ_X1 + 2 * SZ_XN);
  _Float16* kb = (_Float16*)(ws + SZ_X1 + 2 * SZ_XN + SZ_Y);
  _Float16* vb = (_Float16*)(ws + SZ_X1 + 2 * SZ_XN + 2 * SZ_Y);
  _Float16* hb = (_Float16*)(ws + SZ_X1 + 2 * SZ_XN + 3 * SZ_Y);
  _Float16* h2 = (_Float16*)(ws + SZ_X1 + 2 * SZ_XN + 3 * SZ_Y + SZ_H);

  // 1. LPU
  lpu_kernel<<<BQ * HH, 256, 0, stream>>>(x, pdw, pdb, x1);
  // 2. LN1 -> xn (f16)
  ln_kernel<<<BQ * NTOK / 8, 256, 0, stream>>>(x1, ln1g, ln1b, xn);
  // 3. SR conv + BN -> yb
  srbn_kernel<<<BQ * 28, 256, 0, stream>>>(xn, srw, srb, sbg, sbb, sbm, sbv, yb);
  // 4. Q/K/V projections (WMMA)
  gemm_kernel<EP_F16, false><<<dim3(196, 4), 256, 0, stream>>>(
      xn, qw, qbi, nullptr, nullptr, nullptr, nullptr, nullptr, qb, 256, 256);
  gemm_kernel<EP_F16, false><<<dim3(49, 4), 256, 0, stream>>>(
      yb, kw, kbi, nullptr, nullptr, nullptr, nullptr, nullptr, kb, 256, 256);
  gemm_kernel<EP_F16, false><<<dim3(49, 4), 256, 0, stream>>>(
      yb, vw, vbi, nullptr, nullptr, nullptr, nullptr, nullptr, vb, 256, 256);
  // 5. attention (o aliases qb; safe: per-WG read region == write region)
  attn_kernel<<<dim3(196, 4, BQ), 32, 0, stream>>>(qb, kb, vb, rp, qb);
  // 6. out projection + residual -> x2 (aliases x1, in-place per-element)
  gemm_kernel<EP_RESID_F32, false><<<dim3(196, 4), 256, 0, stream>>>(
      qb, aow, aob, x1, nullptr, nullptr, nullptr, nullptr, x1, 256, 256);
  // 7. LN2 -> xn2 (aliases xn)
  ln_kernel<<<BQ * NTOK / 8, 256, 0, stream>>>(x1, ln2g, ln2b, xn);
  // 8. MLP up 256->1024, gelu + BN1 -> hb
  gemm_kernel<EP_GELU_BN_F16, true><<<dim3(196, 16), 256, 0, stream>>>(
      xn, c1w, c1b, nullptr, bn1g, bn1b, bn1m, bn1v, hb, 256, 1024);
  // 9. MLP depthwise + residual + gelu + BN2 -> h2
  dwmlp_kernel<<<BQ * HH * 4, 256, 0, stream>>>(hb, dww, dwb, bn2g, bn2b, bn2m,
                                                bn2v, h2);
  // 10. MLP down 1024->256, BN3 + residual -> d_out (fp32)
  gemm_kernel<EP_BN_RESID_F32, true><<<dim3(196, 4), 256, 0, stream>>>(
      h2, c2w, c2b, x1, bn3g, bn3b, bn3m, bn3v, (float*)d_out, 1024, 256);
}
